// GATLayer_16080357556648
// MI455X (gfx1250) — matrix operand order repaired and checked by reading, not compiled, not run on hardware
//
#include <hip/hip_runtime.h>
#include <hip/hip_bf16.h>
#include <stdint.h>

// ---------------------------------------------------------------------------
// GAT layer, fused flash-softmax formulation for MI455X (gfx1250, wave32).
//   K1: wh = h @ w^T (bf16 WMMA, f32 accum), emit wh1, wh2, wh^T(bf16)
//   K2: flash loop over adj: online softmax + P@wh via bf16 WMMA (j split x4)
//       - adj streamed once with non-temporal loads (256 MB > 192 MB L2)
//       - whT kept temporal (2 MB, reused by all 512 row tiles -> L2 resident)
//       - per-row rescale via wave shuffles, skipped when max unchanged
//   K3: combine j-chunks, normalize, ELU
// ---------------------------------------------------------------------------

#define N_NODES 8192
#define IN_F    512
#define OUT_F   128
#define NEGV    (-1e11f)
#define SLOPE   0.01f
#define JCHUNK  4
#define TILES_M (N_NODES / 16)   // 512

typedef __bf16 bf16_t;
typedef __attribute__((ext_vector_type(16))) __bf16 v16bf;
typedef __attribute__((ext_vector_type(8)))  float  v8f;
typedef __attribute__((ext_vector_type(4)))  int    v4i;
typedef __attribute__((ext_vector_type(4)))  float  v4f;

// ---------------------------------------------------------------------------
// Kernel 1: one block = 16 rows of h; 8 waves each compute one 16x16 n-tile.
// A operand (16x32 bf16): lane = row (m = lane&15), K half by lane>=16.
// B operand (32x16 bf16): lane = col (n = lane&15), K = {kb..kb+7, kb+16..kb+23}, same split as A.
// ---------------------------------------------------------------------------
__global__ __launch_bounds__(256)
void gat_wh_kernel(const float* __restrict__ h, const float* __restrict__ w,
                   const float* __restrict__ a1v, const float* __restrict__ a2v,
                   float* __restrict__ wh1, float* __restrict__ wh2,
                   bf16_t* __restrict__ whT)
{
    __shared__ float lds[16][OUT_F + 4];

    const int tid   = threadIdx.x;
    const int lane  = tid & 31;
    const int wid   = tid >> 5;
    const int nl    = lane & 15;
    const int hi    = lane >> 4;          // 0 or 1
    const int mbase = blockIdx.x * 16;
    const int nbase = wid * 16;

    const float* arow = h + (size_t)(mbase + nl) * IN_F;  // this lane's A row
    const float* wrow = w + (size_t)(nbase + nl) * IN_F;  // this lane's B col (w row)

    v8f c = {0.f, 0.f, 0.f, 0.f, 0.f, 0.f, 0.f, 0.f};

    for (int k0 = 0; k0 < IN_F; k0 += 32) {
        // A: lane holds K = {ka..ka+7, ka+16..ka+23}, ka = k0 + hi*8
        const int ka = k0 + hi * 8;
        v4f f0 = *(const v4f*)(arow + ka);
        v4f f1 = *(const v4f*)(arow + ka + 4);
        v4f f2 = *(const v4f*)(arow + ka + 16);
        v4f f3 = *(const v4f*)(arow + ka + 20);
        v16bf av;
        av[0]  = (bf16_t)f0.x; av[1]  = (bf16_t)f0.y; av[2]  = (bf16_t)f0.z; av[3]  = (bf16_t)f0.w;
        av[4]  = (bf16_t)f1.x; av[5]  = (bf16_t)f1.y; av[6]  = (bf16_t)f1.z; av[7]  = (bf16_t)f1.w;
        av[8]  = (bf16_t)f2.x; av[9]  = (bf16_t)f2.y; av[10] = (bf16_t)f2.z; av[11] = (bf16_t)f2.w;
        av[12] = (bf16_t)f3.x; av[13] = (bf16_t)f3.y; av[14] = (bf16_t)f3.z; av[15] = (bf16_t)f3.w;

        // B: lane holds K = {kb..kb+7, kb+16..kb+23}, kb = k0 + hi*8
        const int kb = k0 + hi * 8;
        v4f g0 = *(const v4f*)(wrow + kb);
        v4f g1 = *(const v4f*)(wrow + kb + 4);
        v4f g2 = *(const v4f*)(wrow + kb + 16);
        v4f g3 = *(const v4f*)(wrow + kb + 20);
        v16bf bv;
        bv[0]  = (bf16_t)g0.x; bv[1]  = (bf16_t)g0.y; bv[2]  = (bf16_t)g0.z; bv[3]  = (bf16_t)g0.w;
        bv[4]  = (bf16_t)g1.x; bv[5]  = (bf16_t)g1.y; bv[6]  = (bf16_t)g1.z; bv[7]  = (bf16_t)g1.w;
        bv[8]  = (bf16_t)g2.x; bv[9]  = (bf16_t)g2.y; bv[10] = (bf16_t)g2.z; bv[11] = (bf16_t)g2.w;
        bv[12] = (bf16_t)g3.x; bv[13] = (bf16_t)g3.y; bv[14] = (bf16_t)g3.z; bv[15] = (bf16_t)g3.w;

        c = __builtin_amdgcn_wmma_f32_16x16x32_bf16(false, av, false, bv,
                                                    (short)0, c, false, false);
    }

    // C layout: VGPR r -> (m = r + 8*hi, n = nbase + (lane&15))
#pragma unroll
    for (int r = 0; r < 8; ++r)
        lds[hi * 8 + r][nbase + nl] = c[r];
    __syncthreads();

    // wh^T in bf16: whT[n][mbase+m]
    for (int e = tid; e < 16 * OUT_F; e += 256) {
        int m = e & 15;
        int n = e >> 4;
        whT[(size_t)n * N_NODES + mbase + m] = (bf16_t)lds[m][n];
    }

    // wh1 / wh2 (row dots with a1 / a2)
    if (tid < 16) {
        float s = 0.f;
#pragma unroll 8
        for (int n = 0; n < OUT_F; ++n) s += lds[tid][n] * a1v[n];
        wh1[mbase + tid] = s;
    } else if (tid < 32) {
        float s = 0.f;
#pragma unroll 8
        for (int n = 0; n < OUT_F; ++n) s += lds[tid - 16][n] * a2v[n];
        wh2[mbase + tid - 16] = s;
    }
}

// ---------------------------------------------------------------------------
// Kernel 2: flash softmax + P@wh. Block = 4 independent waves: same 16-row
// tile, j-chunks 0..3. Wave-private: 8 accumulator tiles (16x16 f32 = v8f).
// Scores are produced directly as the WMMA A operand (contiguous k order, same as B):
//   lane (l&15)=row m, K set = {koff..koff+15}, koff = 16*(l>>4)
// No LDS, no barriers: per-row broadcasts use wave32 shuffles.
// ---------------------------------------------------------------------------
__global__ __launch_bounds__(128)
void gat_attn_kernel(const int* __restrict__ adj,
                     const float* __restrict__ wh1, const float* __restrict__ wh2,
                     const bf16_t* __restrict__ whT,
                     float* __restrict__ cacc, float* __restrict__ cmax,
                     float* __restrict__ csum)
{
    const int tid   = threadIdx.x;
    const int lane  = tid & 31;
    const int wid   = tid >> 5;
    const int mloc  = lane & 15;
    const int hi    = lane >> 4;
    const int koff  = hi * 16;

    const int tile  = blockIdx.x;          // row tile
    const int chunk = wid;                 // j chunk (4 waves = 4 chunks)
    const int mbase = tile * 16;
    const int jlo   = chunk * (N_NODES / JCHUNK);
    const int jhi   = jlo + (N_NODES / JCHUNK);

    const float wh1v = wh1[mbase + mloc];
    const int* arow0 = adj + (size_t)(mbase + mloc) * N_NODES;

    float run_max = -3.0e38f;
    float run_sum = 0.f;
    v8f acc[8];
#pragma unroll
    for (int nt = 0; nt < 8; ++nt)
        acc[nt] = (v8f){0.f, 0.f, 0.f, 0.f, 0.f, 0.f, 0.f, 0.f};

    for (int j = jlo; j < jhi; j += 32) {
        // adj is streamed exactly once -> non-temporal (keep whT in L2)
        const int* arow = arow0 + j + koff;
        v4i q0 = __builtin_nontemporal_load((const v4i*)(arow));
        v4i q1 = __builtin_nontemporal_load((const v4i*)(arow + 4));
        v4i q2 = __builtin_nontemporal_load((const v4i*)(arow + 8));
        v4i q3 = __builtin_nontemporal_load((const v4i*)(arow + 12));
        __builtin_prefetch(arow + 128, 0, 1);   // 4 j-steps ahead

        v4f u0 = *(const v4f*)(wh2 + j + koff);
        v4f u1 = *(const v4f*)(wh2 + j + koff + 4);
        v4f u2 = *(const v4f*)(wh2 + j + koff + 8);
        v4f u3 = *(const v4f*)(wh2 + j + koff + 12);

        float sv[16];
#define GAT_SC(av_, wv_, ix_)                                   \
        do {                                                    \
            float e_ = wh1v + (wv_);                            \
            e_ = e_ > 0.f ? e_ : SLOPE * e_;                    \
            sv[ix_] = ((av_) > 0) ? e_ : NEGV;                  \
        } while (0)
        GAT_SC(q0.x, u0.x, 0);  GAT_SC(q0.y, u0.y, 1);
        GAT_SC(q0.z, u0.z, 2);  GAT_SC(q0.w, u0.w, 3);
        GAT_SC(q1.x, u1.x, 4);  GAT_SC(q1.y, u1.y, 5);
        GAT_SC(q1.z, u1.z, 6);  GAT_SC(q1.w, u1.w, 7);
        GAT_SC(q2.x, u2.x, 8);  GAT_SC(q2.y, u2.y, 9);
        GAT_SC(q2.z, u2.z, 10); GAT_SC(q2.w, u2.w, 11);
        GAT_SC(q3.x, u3.x, 12); GAT_SC(q3.y, u3.y, 13);
        GAT_SC(q3.z, u3.z, 14); GAT_SC(q3.w, u3.w, 15);
#undef GAT_SC

        // per-row tile max (lane pair l, l^16 holds the two K halves of row m)
        float tmax = sv[0];
#pragma unroll
        for (int t = 1; t < 16; ++t) tmax = fmaxf(tmax, sv[t]);
        tmax = fmaxf(tmax, __shfl_xor(tmax, 16, 32));

        // wave-uniform test: does any row's max grow? (rare after warmup)
        const bool need_rescale = (__ballot(tmax > run_max) != 0ull);
        float nmax = fmaxf(run_max, tmax);

        float lsum = 0.f;
        v16bf pa;
#pragma unroll
        for (int t = 0; t < 16; ++t) {
            float p = __expf(sv[t] - nmax);
            lsum += p;
            pa[t] = (bf16_t)p;
        }
        lsum += __shfl_xor(lsum, 16, 32);

        if (need_rescale) {
            const float scale = __expf(run_max - nmax);  // this lane = row mloc
            run_sum = run_sum * scale + lsum;
            // broadcast row scales into the C layout (rows hi*8 + r):
            // lane (hi*8 + r) holds row (hi*8 + r)'s scale
#pragma unroll
            for (int r = 0; r < 8; ++r) {
                const float rs = __shfl(scale, hi * 8 + r, 32);
#pragma unroll
                for (int nt = 0; nt < 8; ++nt) acc[nt][r] *= rs;
            }
            run_max = nmax;
        } else {
            run_sum += lsum;
        }

        // P (16x32) @ wh (32x128): 8 bf16 WMMAs. B from wh^T: lane = col n,
        // 16 contiguous K values -> single 32B load (temporal, L2-resident).
#pragma unroll
        for (int nt = 0; nt < 8; ++nt) {
            v16bf bb = *(const v16bf*)(whT + (size_t)(nt * 16 + mloc) * N_NODES
                                           + j + hi * 16);
            acc[nt] = __builtin_amdgcn_wmma_f32_16x16x32_bf16(
                          false, pa, false, bb, (short)0, acc[nt], false, false);
        }
    }

    // write partial flash state for this (chunk, tile); acc is write-once ->
    // non-temporal stores (don't let 16 MB of partials churn L2)
    const size_t sb = (size_t)(chunk * TILES_M + tile) * 16;
    if (hi == 0) {
        cmax[sb + mloc] = run_max;
        csum[sb + mloc] = run_sum;
    }
#pragma unroll
    for (int nt = 0; nt < 8; ++nt) {
#pragma unroll
        for (int r = 0; r < 8; ++r)
            __builtin_nontemporal_store(acc[nt][r],
                &cacc[(sb + hi * 8 + r) * OUT_F + nt * 16 + mloc]);
    }
}

// ---------------------------------------------------------------------------
// Kernel 3: combine the JCHUNK partials per row, normalize, ELU.
// One block per row tile; thread = output column.
// ---------------------------------------------------------------------------
__global__ __launch_bounds__(128)
void gat_finish_kernel(const float* __restrict__ cacc, const float* __restrict__ cmax,
                       const float* __restrict__ csum, float* __restrict__ out)
{
    const int tile = blockIdx.x;
    const int col  = threadIdx.x;

    for (int m = 0; m < 16; ++m) {
        float cm[JCHUNK];
        float gmax = -3.0e38f;
#pragma unroll
        for (int c = 0; c < JCHUNK; ++c) {
            cm[c] = cmax[(size_t)(c * TILES_M + tile) * 16 + m];
            gmax = fmaxf(gmax, cm[c]);
        }
        float gsum = 0.f, f[JCHUNK];
#pragma unroll
        for (int c = 0; c < JCHUNK; ++c) {
            f[c] = __expf(cm[c] - gmax);
            gsum += csum[(size_t)(c * TILES_M + tile) * 16 + m] * f[c];
        }
        float x = 0.f;
#pragma unroll
        for (int c = 0; c < JCHUNK; ++c)
            x += cacc[((size_t)(c * TILES_M + tile) * 16 + m) * OUT_F + col] * f[c];
        x /= gsum;
        x = x > 0.f ? x : (__expf(x) - 1.f);   // ELU (alpha = 1)
        out[(size_t)(tile * 16 + m) * OUT_F + col] = x;
    }
}

// ---------------------------------------------------------------------------
extern "C" void kernel_launch(void* const* d_in, const int* in_sizes, int n_in,
                              void* d_out, int out_size, void* d_ws, size_t ws_size,
                              hipStream_t stream)
{
    (void)in_sizes; (void)n_in; (void)out_size; (void)ws_size;

    const float* h   = (const float*)d_in[0];
    const int*   adj = (const int*)  d_in[1];
    const float* w   = (const float*)d_in[2];
    const float* a1v = (const float*)d_in[3];
    const float* a2v = (const float*)d_in[4];
    float* out = (float*)d_out;

    char* ws = (char*)d_ws;
    float*  wh1  = (float*)ws;                                   // 32 KB
    float*  wh2  = (float*)(ws + 32 * 1024);                     // 32 KB
    bf16_t* whT  = (bf16_t*)(ws + 64 * 1024);                    // 2 MB
    float*  cacc = (float*)(ws + 64 * 1024 + 2u * 1024 * 1024);  // 16 MB
    float*  cmax = (float*)(ws + 64 * 1024 + 18u * 1024 * 1024); // 128 KB
    float*  csum = (float*)(ws + 64 * 1024 + 18u * 1024 * 1024 + 128 * 1024);

    gat_wh_kernel<<<TILES_M, 256, 0, stream>>>(h, w, a1v, a2v, wh1, wh2, whT);
    gat_attn_kernel<<<TILES_M, 128, 0, stream>>>(adj, wh1, wh2, whT,
                                                 cacc, cmax, csum);
    gat_finish_kernel<<<TILES_M, 128, 0, stream>>>(cacc, cmax, csum, out);
}
